// LearnableQuantizedLinear_49658411876685
// MI455X (gfx1250) — matrix operand order repaired and checked by reading, not compiled, not run on hardware
//
#include <hip/hip_runtime.h>
#include <hip/hip_bf16.h>
#include <stdint.h>

// Problem dims (fixed by the reference)
constexpr int M_DIM = 8192;   // batch*seq
constexpr int N_DIM = 4096;   // out_features
constexpr int K_DIM = 4096;   // in_features

typedef __bf16 bf16_t;
typedef __attribute__((ext_vector_type(16))) __bf16 bf16x16;
typedef __attribute__((ext_vector_type(8)))  float  f32x8;

// ---------------------------------------------------------------------------
// float -> bf16 round-to-nearest-even
// ---------------------------------------------------------------------------
__device__ __forceinline__ unsigned short f2bf(float f) {
    union { float f; unsigned u; } x; x.f = f;
    unsigned r = x.u + 0x7FFFu + ((x.u >> 16) & 1u);
    return (unsigned short)(r >> 16);
}

// ---------------------------------------------------------------------------
// CDNA5 async global->LDS copy (16B per lane), tracked by ASYNCcnt.
// lds_addr: low 32 bits of a shared-memory pointer (LDS byte offset).
// gbase:    64-bit SGPR base (kernel-arg pointer), byte_off: per-lane i32.
// ---------------------------------------------------------------------------
__device__ __forceinline__ void async_copy_16B(unsigned lds_addr,
                                               const void* gbase,
                                               unsigned byte_off) {
    asm volatile("global_load_async_to_lds_b128 %0, %1, %2"
                 :
                 : "v"(lds_addr), "v"(byte_off), "s"(gbase)
                 : "memory");
}
__device__ __forceinline__ void wait_asynccnt0() {
    asm volatile("s_wait_asynccnt 0" ::: "memory");
}

// ---------------------------------------------------------------------------
// Kernel 1: NF4 quantize->dequantize weight into bf16 (memory-bound prepass)
// ---------------------------------------------------------------------------
__global__ void __launch_bounds__(256)
nf4_dequant_kernel(const float* __restrict__ w,
                   const float* __restrict__ scales,
                   bf16_t* __restrict__ qw) {
    int i = (blockIdx.x * 256 + threadIdx.x) * 4;       // 4 elems / thread
    float4 wv = *(const float4*)(w + i);
    float scale = scales[i >> 6];                        // 4 | 64 -> same group
    float inv   = 1.0f / scale;

    float in[4] = { wv.x, wv.y, wv.z, wv.w };
    unsigned short h[4];
#pragma unroll
    for (int e = 0; e < 4; ++e) {
        float s = in[e] * inv;
        s = fminf(1.0f, fmaxf(-1.0f, s));
        float v = -1.0f;
        v = (s >= -0.84809640f) ? -0.69619280f : v;
        v = (s >= -0.61063293f) ? -0.52507305f : v;
        v = (s >= -0.45999527f) ? -0.39491749f : v;
        v = (s >= -0.33967944f) ? -0.28444138f : v;
        v = (s >= -0.23460741f) ? -0.18477343f : v;
        v = (s >= -0.13791173f) ? -0.09105004f : v;
        v = (s >= -0.04552502f) ?  0.00000000f : v;
        v = (s >=  0.03979015f) ?  0.07958030f : v;
        v = (s >=  0.12025525f) ?  0.16093020f : v;
        v = (s >=  0.20352130f) ?  0.24611239f : v;
        v = (s >=  0.29201382f) ?  0.33791524f : v;
        v = (s >=  0.38931254f) ?  0.44070983f : v;
        v = (s >=  0.50166341f) ?  0.56261700f : v;
        v = (s >=  0.64278692f) ?  0.72295684f : v;
        v = (s >=  0.86147842f) ?  1.00000000f : v;
        h[e] = f2bf(v * scale);
    }
    uint2 pk;
    pk.x = (unsigned)h[0] | ((unsigned)h[1] << 16);
    pk.y = (unsigned)h[2] | ((unsigned)h[3] << 16);
    *(uint2*)(qw + i) = pk;
}

// ---------------------------------------------------------------------------
// Kernel 2: activations f32 -> bf16 (memory-bound prepass)
// ---------------------------------------------------------------------------
__global__ void __launch_bounds__(256)
f32_to_bf16_kernel(const float* __restrict__ in, bf16_t* __restrict__ out) {
    int i = (blockIdx.x * 256 + threadIdx.x) * 8;       // 8 elems / thread
    float4 a = *(const float4*)(in + i);
    float4 b = *(const float4*)(in + i + 4);
    uint4 pk;
    pk.x = (unsigned)f2bf(a.x) | ((unsigned)f2bf(a.y) << 16);
    pk.y = (unsigned)f2bf(a.z) | ((unsigned)f2bf(a.w) << 16);
    pk.z = (unsigned)f2bf(b.x) | ((unsigned)f2bf(b.y) << 16);
    pk.w = (unsigned)f2bf(b.z) | ((unsigned)f2bf(b.w) << 16);
    *(uint4*)(out + i) = pk;
}

// ---------------------------------------------------------------------------
// Kernel 3: bf16 WMMA GEMM.  C[M,N] = A[M,K] * B[N,K]^T, f32 accumulate.
// Block tile 128x128, 8 waves (wave32), wave tile 64x32 = 4x2 WMMA accums.
// Double-buffered LDS fed by global_load_async_to_lds_b128 (ASYNCcnt),
// so no VGPR staging of tiles -> no scratch spills.
// ---------------------------------------------------------------------------
constexpr int BM  = 128;
constexpr int BN  = 128;
constexpr int BK  = 32;
constexpr int LDT = BK + 8;   // padded LDS row stride (halfs) -> no bank conflicts

__global__ void __launch_bounds__(256)
nf4_gemm_wmma(const bf16_t* __restrict__ A,    // [M, K] bf16
              const bf16_t* __restrict__ B,    // [N, K] bf16
              float* __restrict__ C) {         // [M, N] f32
    __shared__ bf16_t As[2][BM * LDT];
    __shared__ bf16_t Bs[2][BN * LDT];

    const int tid   = threadIdx.x;
    const int lane  = tid & 31;         // wave32
    const int wave  = tid >> 5;         // 0..7
    const int wm    = wave & 1;         // wave row  -> *64
    const int wn    = wave >> 1;        // wave col  -> *32
    const int row16 = lane & 15;
    const int hi    = lane >> 4;

    const int bm = blockIdx.y * BM;
    const int bn = blockIdx.x * BN;

    f32x8 acc[4][2];
#pragma unroll
    for (int mt = 0; mt < 4; ++mt)
#pragma unroll
        for (int nt = 0; nt < 2; ++nt)
#pragma unroll
            for (int j = 0; j < 8; ++j) acc[mt][nt][j] = 0.0f;

    // ----- tile geometry for async copies:
    // 128 rows x 32 halfs = 64 B/row = 4 x 16B chunks; 512 chunks / 256 thr = 2.
    const int lrow = tid >> 2;          // 0..63  (row within pass half)
    const int lcol = tid & 3;           // chunk 0..3
    // per-lane global byte offsets into this block's tiles (k-invariant part)
    const unsigned a_goff0 = (unsigned)((bm + lrow) * K_DIM + lcol * 8) * 2u;
    const unsigned a_goff1 = (unsigned)((bm + 64 + lrow) * K_DIM + lcol * 8) * 2u;
    const unsigned b_goff0 = (unsigned)((bn + lrow) * K_DIM + lcol * 8) * 2u;
    const unsigned b_goff1 = (unsigned)((bn + 64 + lrow) * K_DIM + lcol * 8) * 2u;
    // per-lane LDS byte addresses (low 32 bits of shared pointer = LDS offset)
    const unsigned a_loff0 = (unsigned)(uintptr_t)&As[0][lrow * LDT + lcol * 8];
    const unsigned a_loff1 = (unsigned)(uintptr_t)&As[0][(64 + lrow) * LDT + lcol * 8];
    const unsigned b_loff0 = (unsigned)(uintptr_t)&Bs[0][lrow * LDT + lcol * 8];
    const unsigned b_loff1 = (unsigned)(uintptr_t)&Bs[0][(64 + lrow) * LDT + lcol * 8];
    constexpr unsigned LDS_BUF = (unsigned)(BM * LDT * sizeof(bf16_t)); // buf stride

    auto issue_async = [&](int kt, int buf) {
        const unsigned kb = (unsigned)(kt * BK * 2);    // k byte offset
        const unsigned lb = buf ? LDS_BUF : 0u;
        async_copy_16B(a_loff0 + lb, A, a_goff0 + kb);
        async_copy_16B(a_loff1 + lb, A, a_goff1 + kb);
        async_copy_16B(b_loff0 + lb, B, b_goff0 + kb);
        async_copy_16B(b_loff1 + lb, B, b_goff1 + kb);
    };

    union Frag { bf16x16 v; uint4 q[2]; };

    auto compute = [&](int buf) {
        Frag af[4], bf[2];
        // A fragment: lane = row M, K split across VGPR halves (16-bit A layout)
#pragma unroll
        for (int mt = 0; mt < 4; ++mt) {
            const bf16_t* base = &As[buf][(wm * 64 + mt * 16 + row16) * LDT];
            af[mt].q[0] = *(const uint4*)(base + hi * 8);
            af[mt].q[1] = *(const uint4*)(base + 16 + hi * 8);
        }
        // B fragment: lane = column N, K split across VGPR halves (same as A)
#pragma unroll
        for (int nt = 0; nt < 2; ++nt) {
            const bf16_t* base = &Bs[buf][(wn * 32 + nt * 16 + row16) * LDT + hi * 8];
            bf[nt].q[0] = *(const uint4*)(base);
            bf[nt].q[1] = *(const uint4*)(base + 16);
        }
#pragma unroll
        for (int mt = 0; mt < 4; ++mt)
#pragma unroll
            for (int nt = 0; nt < 2; ++nt)
                acc[mt][nt] = __builtin_amdgcn_wmma_f32_16x16x32_bf16(
                    /*neg_a=*/false, af[mt].v,
                    /*neg_b=*/false, bf[nt].v,
                    /*c_mod=*/(short)0, acc[mt][nt],
                    /*reuse_a=*/false, /*reuse_b=*/false);
    };

    constexpr int NT = K_DIM / BK;   // 128 stages
    issue_async(0, 0);
    wait_asynccnt0();
    __syncthreads();

    for (int kt = 0; kt < NT; ++kt) {
        const int cur = kt & 1;
        if (kt + 1 < NT) issue_async(kt + 1, cur ^ 1);   // DMA next tile -> LDS
        if (kt + 2 < NT) {                               // L2 prefetch 2 ahead
            __builtin_prefetch(A + (size_t)(bm + lrow) * K_DIM + (kt + 2) * BK, 0, 1);
            __builtin_prefetch(B + (size_t)(bn + lrow) * K_DIM + (kt + 2) * BK, 0, 1);
        }
        compute(cur);                                    // ds_load + v_wmma
        if (kt + 1 < NT) wait_asynccnt0();               // this wave's DMAs done
        __syncthreads();                                 // all waves' DMAs done
    }

    // ----- write back: 16x16 f32 C layout -> M = r + 8*hi, N = lane&15
#pragma unroll
    for (int mt = 0; mt < 4; ++mt)
#pragma unroll
        for (int nt = 0; nt < 2; ++nt)
#pragma unroll
            for (int r = 0; r < 8; ++r) {
                int m = bm + wm * 64 + mt * 16 + r + hi * 8;
                int n = bn + wn * 32 + nt * 16 + row16;
                C[(size_t)m * N_DIM + n] = acc[mt][nt][r];
            }
}

// ---------------------------------------------------------------------------
extern "C" void kernel_launch(void* const* d_in, const int* in_sizes, int n_in,
                              void* d_out, int out_size, void* d_ws, size_t ws_size,
                              hipStream_t stream) {
    const float* inp    = (const float*)d_in[0];   // [4,2048,4096]
    const float* weight = (const float*)d_in[1];   // [4096,4096]
    const float* scales = (const float*)d_in[2];   // [262144]
    float* out = (float*)d_out;

    bf16_t* qw = (bf16_t*)d_ws;                                        // 32 MB
    bf16_t* ax = (bf16_t*)((char*)d_ws + (size_t)N_DIM * K_DIM * 2);   // 64 MB

    nf4_dequant_kernel<<<(N_DIM * K_DIM) / (256 * 4), 256, 0, stream>>>(weight, scales, qw);
    f32_to_bf16_kernel<<<(M_DIM * K_DIM) / (256 * 8), 256, 0, stream>>>(inp, ax);

    dim3 grid(N_DIM / BN, M_DIM / BM);   // (32, 64)
    nf4_gemm_wmma<<<grid, 256, 0, stream>>>(ax, qw, out);
}
